// ProcessNeurons_30262339568313
// MI455X (gfx1250) — matrix operand order repaired and checked by reading, not compiled, not run on hardware
//
#include <hip/hip_runtime.h>
#include <hip/hip_bf16.h>
#include <stdint.h>

// Problem shape (fixed by the reference)
#define B_DIM  4
#define S_DIM  1024
#define H_DIM  512
#define NIN    64
#define NPROC  32
#define R_DIM  (B_DIM * S_DIM)   // 4096 GEMM rows

typedef __attribute__((ext_vector_type(16))) __bf16 v16bf;
typedef __attribute__((ext_vector_type(8)))  float  v8f;
typedef __attribute__((ext_vector_type(4)))  int    v4i;

#if __has_builtin(__builtin_amdgcn_global_load_async_to_lds_b128)
#define HAVE_ASYNC 1
#else
#define HAVE_ASYNC 0
#endif

#if __has_builtin(__builtin_amdgcn_s_wait_asynccnt)
#define WAIT_ASYNC(n) __builtin_amdgcn_s_wait_asynccnt(n)
#else
#define WAIT_ASYNC(n) asm volatile("s_wait_asynccnt %0" ::"i"(n))
#endif

typedef __attribute__((address_space(1))) v4i gv4i;   // global int4 (AS1)
typedef __attribute__((address_space(3))) v4i lv4i;   // LDS int4 (AS3)

static __device__ __forceinline__ uint32_t f32_to_bf16(float f) {
  uint32_t u = __float_as_uint(f);
  uint32_t r = u + 0x7FFFu + ((u >> 16) & 1u);   // round-to-nearest-even
  return r >> 16;
}
static __device__ __forceinline__ uint32_t pack_bf16x2(float a, float b) {
  return f32_to_bf16(a) | (f32_to_bf16(b) << 16);
}

// ---------------------------------------------------------------------------
// Kernel 0: one-time f32 -> bf16 conversion of transforms into workspace
// ---------------------------------------------------------------------------
__global__ __launch_bounds__(256) void convert_T_bf16(const float* __restrict__ src,
                                                      uint16_t* __restrict__ dst,
                                                      int n4) {
  int i = blockIdx.x * 256 + threadIdx.x;
  if (i < n4) {
    float4 v = ((const float4*)src)[i];
    uint2 o;
    o.x = pack_bf16x2(v.x, v.y);
    o.y = pack_bf16x2(v.z, v.w);
    ((uint2*)dst)[i] = o;
  }
}

// ---------------------------------------------------------------------------
// Kernel 1: 1D conv (kernel 5, pad 2, full reduction over NIN) + sigmoid gate
// ---------------------------------------------------------------------------
__global__ __launch_bounds__(256) void conv_sigmoid(const float* __restrict__ IA,
                                                    const float* __restrict__ W,
                                                    const float* __restrict__ Bv,
                                                    float* __restrict__ P) {
  __shared__ float acts[12 * NIN];          // rows s0-2 .. s0+9
  const int tid  = threadIdx.x;
  const int pos0 = blockIdx.x * 8;          // 8 | S_DIM, never crosses batch
  const int b    = pos0 / S_DIM;
  const int s0   = pos0 % S_DIM;

  for (int i = tid; i < 12 * NIN; i += 256) {
    int rr = i >> 6, c = i & 63;
    int s  = s0 + rr - 2;
    acts[i] = (s >= 0 && s < S_DIM) ? IA[((size_t)(b * S_DIM + s)) * NIN + c] : 0.0f;
  }
  __syncthreads();

  const int si = tid >> 5;
  const int n  = tid & 31;
  float sum = Bv[n];
#pragma unroll
  for (int k = 0; k < 5; ++k) {
    const float* arow = &acts[(si + k) * NIN];
    const float* wrow = &W[((n * 5) + k) * NIN];
#pragma unroll
    for (int i = 0; i < NIN; i += 4) {
      float4 a = *(const float4*)&arow[i];
      float4 w = *(const float4*)&wrow[i];
      sum = fmaf(a.x, w.x, sum);
      sum = fmaf(a.y, w.y, sum);
      sum = fmaf(a.z, w.z, sum);
      sum = fmaf(a.w, w.w, sum);
    }
  }
  float p = 1.0f / (1.0f + expf(-sum));
  P[((size_t)(b * S_DIM + s0 + si)) * NPROC + n] = p;
}

// ---------------------------------------------------------------------------
// Shared GEMM tiling constants
// WG tile 128x128, 8 waves (4 wm x 2 wn), wave tile 32x64 (MT=2, NT=4)
// ---------------------------------------------------------------------------
#define M_BLK  128
#define N_BLK  128
#define XS_STR 520   // 512 + 8 bf16 pad
#define BS_STR 136   // 128 + 8 bf16 pad
#define PT_STR 132

constexpr int XS_ELEMS    = M_BLK * XS_STR;                 // bf16
constexpr int BS_ELEMS    = H_DIM * BS_STR;                 // bf16 (sync path: full 512-row panel)
constexpr int CHUNK_ROWS  = 128;                            // async path: K rows per chunk
constexpr int CHUNK_ELEMS = CHUNK_ROWS * BS_STR;            // bf16
constexpr int NCHUNK      = NPROC * (H_DIM / CHUNK_ROWS);   // 128 chunks total

constexpr int SMEM_SYNC  = XS_ELEMS * 2 + BS_ELEMS * 2 + NPROC * PT_STR * 4;        // 289,280 B
constexpr int SMEM_ASYNC = XS_ELEMS * 2 + 3 * CHUNK_ELEMS * 2 + NPROC * PT_STR * 4; // 254,464 B

// ---------------------------------------------------------------------------
// Kernel 2a: sync-staged fused gated GEMM (fallback / small-ws path)
// out[r,d] = sum_n p[r,n] * (x[r,:] @ T[n,:,:])[d]
// ---------------------------------------------------------------------------
template <bool PRE>
__global__ __launch_bounds__(256) void fused_gemm(const float* __restrict__ X,
                                                  const float* __restrict__ P,
                                                  const uint16_t* __restrict__ Tb,
                                                  const float* __restrict__ Tf,
                                                  float* __restrict__ Out) {
  extern __shared__ char smem[];
  uint16_t* xs = (uint16_t*)smem;              // [M_BLK][XS_STR] bf16
  uint16_t* bs = xs + XS_ELEMS;                // [H_DIM][BS_STR] bf16
  float*    pt = (float*)(bs + BS_ELEMS);      // [NPROC][PT_STR] f32 (transposed gates)

  const int tid   = threadIdx.x;
  const int lane  = tid & 31;
  const int wv    = tid >> 5;
  const int wm    = wv & 3;
  const int wn    = wv >> 2;
  const int r0    = blockIdx.x * M_BLK;
  const int d0    = blockIdx.y * N_BLK;
  const int rw    = wm * 32;
  const int cw    = wn * 64;
  const int lhalf = (lane >> 4) & 1;
  const int lmod  = lane & 15;

  for (int c = tid; c < M_BLK * (H_DIM / 4); c += 256) {
    int row = c >> 7;
    int col = (c & 127) * 4;
    float4 v = *(const float4*)&X[((size_t)(r0 + row)) * H_DIM + col];
    uint2 o;
    o.x = pack_bf16x2(v.x, v.y);
    o.y = pack_bf16x2(v.z, v.w);
    *(uint2*)&xs[row * XS_STR + col] = o;
  }
  for (int i = tid; i < M_BLK * NPROC; i += 256) {
    int row = i >> 5, n = i & 31;
    pt[n * PT_STR + row] = P[((size_t)(r0 + row)) * NPROC + n];
  }

  const v8f vzero = {};
  v8f main_acc[2][4];
#pragma unroll
  for (int t = 0; t < 2; ++t)
#pragma unroll
    for (int u = 0; u < 4; ++u) main_acc[t][u] = vzero;

  for (int n = 0; n < NPROC; ++n) {
    __syncthreads();
    if (PRE) {
      const uint16_t* src = Tb + ((size_t)n * H_DIM) * H_DIM + d0;
      for (int c = tid; c < H_DIM * (N_BLK / 8); c += 256) {
        int row = c >> 4;
        int col = (c & 15) * 8;
        uint4 v = *(const uint4*)&src[(size_t)row * H_DIM + col];
        *(uint4*)&bs[row * BS_STR + col] = v;
      }
    } else {
      const float* src = Tf + ((size_t)n * H_DIM) * H_DIM + d0;
      for (int c = tid; c < H_DIM * (N_BLK / 8); c += 256) {
        int row = c >> 4;
        int col = (c & 15) * 8;
        float4 a = *(const float4*)&src[(size_t)row * H_DIM + col];
        float4 b = *(const float4*)&src[(size_t)row * H_DIM + col + 4];
        uint4 o;
        o.x = pack_bf16x2(a.x, a.y);
        o.y = pack_bf16x2(a.z, a.w);
        o.z = pack_bf16x2(b.x, b.y);
        o.w = pack_bf16x2(b.z, b.w);
        *(uint4*)&bs[row * BS_STR + col] = o;
      }
    }
    __syncthreads();

    float4 pa[2], pb[2];
#pragma unroll
    for (int t = 0; t < 2; ++t) {
      const float* pp = &pt[n * PT_STR + rw + t * 16 + lhalf * 8];
      pa[t] = *(const float4*)pp;
      pb[t] = *(const float4*)(pp + 4);
    }

    v8f acc[2][4];
#pragma unroll
    for (int t = 0; t < 2; ++t)
#pragma unroll
      for (int u = 0; u < 4; ++u) acc[t][u] = vzero;

    for (int hc = 0; hc < H_DIM; hc += 32) {
      v16bf afrag[2];
#pragma unroll
      for (int t = 0; t < 2; ++t) {
        const uint16_t* ap = &xs[(rw + t * 16 + lmod) * XS_STR + hc + lhalf * 8];
        union { uint4 q[2]; v16bf v; } au;
        au.q[0] = *(const uint4*)ap;
        au.q[1] = *(const uint4*)(ap + 16);
        afrag[t] = au.v;
      }
      v16bf bfrag[4];
#pragma unroll
      for (int u = 0; u < 4; ++u) {
        const uint16_t* bp = &bs[(hc + lhalf * 8) * BS_STR + cw + u * 16 + lmod];
        union { uint16_t e[16]; v16bf v; } bu;
#pragma unroll
        for (int s = 0; s < 16; ++s) bu.e[s] = bp[((s & 7) + (s >> 3) * 16) * BS_STR];
        bfrag[u] = bu.v;
      }
#pragma unroll
      for (int t = 0; t < 2; ++t)
#pragma unroll
        for (int u = 0; u < 4; ++u)
          acc[t][u] = __builtin_amdgcn_wmma_f32_16x16x32_bf16(
              false, afrag[t], false, bfrag[u], (short)0, acc[t][u], false, false);
    }

#pragma unroll
    for (int t = 0; t < 2; ++t) {
      float pj[8] = {pa[t].x, pa[t].y, pa[t].z, pa[t].w,
                     pb[t].x, pb[t].y, pb[t].z, pb[t].w};
#pragma unroll
      for (int u = 0; u < 4; ++u)
#pragma unroll
        for (int j = 0; j < 8; ++j)
          main_acc[t][u][j] = fmaf(pj[j], acc[t][u][j], main_acc[t][u][j]);
    }
  }

#pragma unroll
  for (int t = 0; t < 2; ++t)
#pragma unroll
    for (int u = 0; u < 4; ++u)
#pragma unroll
      for (int j = 0; j < 8; ++j) {
        int row = r0 + rw + t * 16 + lhalf * 8 + j;
        int col = d0 + cw + u * 16 + lmod;
        Out[(size_t)row * H_DIM + col] = main_acc[t][u][j];
      }
}

#if HAVE_ASYNC
// ---------------------------------------------------------------------------
// Kernel 2b: async-staged, software-pipelined fused gated GEMM.
// 3 LDS chunk buffers; GLOBAL_LOAD_ASYNC_TO_LDS_B128 stages chunk i+2 while
// WMMAs consume chunk i. 8 b128 async loads per lane per chunk (in-order
// completion => s_wait_asynccnt<=8 means "my current chunk landed").
// ---------------------------------------------------------------------------
static __device__ __forceinline__ void issue_chunk_async(const uint16_t* __restrict__ Tb,
                                                         uint16_t* bs_base, int d0, int tid,
                                                         int chunk) {
  const uint16_t* src = Tb +
      ((size_t)(chunk >> 2) * H_DIM + (size_t)((chunk & 3) * CHUNK_ROWS)) * H_DIM + d0;
  uint16_t* dst = bs_base + (chunk % 3) * CHUNK_ELEMS;
#pragma unroll
  for (int j = 0; j < 8; ++j) {
    int s   = tid + 256 * j;          // 2048 16B segments per chunk
    int row = s >> 4;
    int col = (s & 15) * 8;
    __builtin_amdgcn_global_load_async_to_lds_b128(
        (gv4i*)(src + (size_t)row * H_DIM + col),
        (lv4i*)(dst + row * BS_STR + col), 0, 0);
  }
}

__global__ __launch_bounds__(256) void fused_gemm_async(const float* __restrict__ X,
                                                        const float* __restrict__ P,
                                                        const uint16_t* __restrict__ Tb,
                                                        float* __restrict__ Out) {
  extern __shared__ char smem[];
  uint16_t* xs = (uint16_t*)smem;                  // [M_BLK][XS_STR] bf16
  uint16_t* bs = xs + XS_ELEMS;                    // 3 x [CHUNK_ROWS][BS_STR] bf16
  float*    pt = (float*)(bs + 3 * CHUNK_ELEMS);   // [NPROC][PT_STR] f32

  const int tid   = threadIdx.x;
  const int lane  = tid & 31;
  const int wv    = tid >> 5;
  const int wm    = wv & 3;
  const int wn    = wv >> 2;
  const int r0    = blockIdx.x * M_BLK;
  const int d0    = blockIdx.y * N_BLK;
  const int rw    = wm * 32;
  const int cw    = wn * 64;
  const int lhalf = (lane >> 4) & 1;
  const int lmod  = lane & 15;

  // kick off the pipeline before doing the (latency-heavy) x-panel prologue
  issue_chunk_async(Tb, bs, d0, tid, 0);
  issue_chunk_async(Tb, bs, d0, tid, 1);

  for (int c = tid; c < M_BLK * (H_DIM / 4); c += 256) {
    int row = c >> 7;
    int col = (c & 127) * 4;
    float4 v = *(const float4*)&X[((size_t)(r0 + row)) * H_DIM + col];
    uint2 o;
    o.x = pack_bf16x2(v.x, v.y);
    o.y = pack_bf16x2(v.z, v.w);
    *(uint2*)&xs[row * XS_STR + col] = o;
  }
  for (int i = tid; i < M_BLK * NPROC; i += 256) {
    int row = i >> 5, n = i & 31;
    pt[n * PT_STR + row] = P[((size_t)(r0 + row)) * NPROC + n];
  }
  __syncthreads();   // xs/pt visible (async chunks not needed yet)

  const v8f vzero = {};
  v8f main_acc[2][4];
#pragma unroll
  for (int t = 0; t < 2; ++t)
#pragma unroll
    for (int u = 0; u < 4; ++u) main_acc[t][u] = vzero;

  for (int n = 0; n < NPROC; ++n) {
    float4 pa[2], pb[2];
#pragma unroll
    for (int t = 0; t < 2; ++t) {
      const float* pp = &pt[n * PT_STR + rw + t * 16 + lhalf * 8];
      pa[t] = *(const float4*)pp;
      pb[t] = *(const float4*)(pp + 4);
    }

    v8f acc[2][4];
#pragma unroll
    for (int t = 0; t < 2; ++t)
#pragma unroll
      for (int u = 0; u < 4; ++u) acc[t][u] = vzero;

    for (int c = 0; c < 4; ++c) {          // 4 chunks of 128 K-rows per n
      const int i = n * 4 + c;
      // my 8 async loads for chunk i are done once <=8 remain outstanding
      if (i < NCHUNK - 1) { WAIT_ASYNC(8); } else { WAIT_ASYNC(0); }
      __syncthreads();   // all waves' portions visible; buf[(i+2)%3] is free
      if (i + 2 < NCHUNK) issue_chunk_async(Tb, bs, d0, tid, i + 2);

      const uint16_t* buf = bs + (i % 3) * CHUNK_ELEMS;
      const int kc = c * CHUNK_ROWS;       // K offset within xs for this chunk

      for (int h4 = 0; h4 < CHUNK_ROWS; h4 += 32) {
        v16bf afrag[2];
#pragma unroll
        for (int t = 0; t < 2; ++t) {
          const uint16_t* ap = &xs[(rw + t * 16 + lmod) * XS_STR + kc + h4 + lhalf * 8];
          union { uint4 q[2]; v16bf v; } au;
          au.q[0] = *(const uint4*)ap;
          au.q[1] = *(const uint4*)(ap + 16);
          afrag[t] = au.v;
        }
        v16bf bfrag[4];
#pragma unroll
        for (int u = 0; u < 4; ++u) {
          const uint16_t* bp = &buf[(h4 + lhalf * 8) * BS_STR + cw + u * 16 + lmod];
          union { uint16_t e[16]; v16bf v; } bu;
#pragma unroll
          for (int s = 0; s < 16; ++s) bu.e[s] = bp[((s & 7) + (s >> 3) * 16) * BS_STR];
          bfrag[u] = bu.v;
        }
#pragma unroll
        for (int t = 0; t < 2; ++t)
#pragma unroll
          for (int u = 0; u < 4; ++u)
            acc[t][u] = __builtin_amdgcn_wmma_f32_16x16x32_bf16(
                false, afrag[t], false, bfrag[u], (short)0, acc[t][u], false, false);
      }
    }

#pragma unroll
    for (int t = 0; t < 2; ++t) {
      float pj[8] = {pa[t].x, pa[t].y, pa[t].z, pa[t].w,
                     pb[t].x, pb[t].y, pb[t].z, pb[t].w};
#pragma unroll
      for (int u = 0; u < 4; ++u)
#pragma unroll
        for (int j = 0; j < 8; ++j)
          main_acc[t][u][j] = fmaf(pj[j], acc[t][u][j], main_acc[t][u][j]);
    }
  }

#pragma unroll
  for (int t = 0; t < 2; ++t)
#pragma unroll
    for (int u = 0; u < 4; ++u)
#pragma unroll
      for (int j = 0; j < 8; ++j) {
        int row = r0 + rw + t * 16 + lhalf * 8 + j;
        int col = d0 + cw + u * 16 + lmod;
        Out[(size_t)row * H_DIM + col] = main_acc[t][u][j];
      }
}
#endif  // HAVE_ASYNC

// ---------------------------------------------------------------------------
extern "C" void kernel_launch(void* const* d_in, const int* in_sizes, int n_in,
                              void* d_out, int out_size, void* d_ws, size_t ws_size,
                              hipStream_t stream) {
  (void)in_sizes; (void)n_in; (void)out_size;
  const float* X  = (const float*)d_in[0];   // intermediate      [B,S,H]
  const float* IA = (const float*)d_in[1];   // input_activations [B,S,NIN]
  const float* W  = (const float*)d_in[2];   // conv_w            [NPROC,1,5,NIN]
  const float* Bv = (const float*)d_in[3];   // conv_b            [NPROC]
  const float* Tf = (const float*)d_in[4];   // transforms        [NPROC,H,H]

  float* Out = (float*)d_out;                      // [B,S,H]
  float* P   = Out + (size_t)R_DIM * H_DIM;        // [B,S,NPROC] (2nd output)

  conv_sigmoid<<<R_DIM / 8, 256, 0, stream>>>(IA, W, Bv, P);

  dim3 grid(R_DIM / M_BLK, H_DIM / N_BLK);   // 32 x 4
  const size_t tb_bytes = (size_t)NPROC * H_DIM * H_DIM * sizeof(uint16_t);

  if (ws_size >= tb_bytes) {
    uint16_t* Tb = (uint16_t*)d_ws;
    int n4 = (NPROC * H_DIM * H_DIM) / 4;
    convert_T_bf16<<<(n4 + 255) / 256, 256, 0, stream>>>(Tf, Tb, n4);
#if HAVE_ASYNC
    (void)hipFuncSetAttribute((const void*)fused_gemm_async,
                              hipFuncAttributeMaxDynamicSharedMemorySize, SMEM_ASYNC);
    fused_gemm_async<<<grid, 256, SMEM_ASYNC, stream>>>(X, P, Tb, Out);
#else
    (void)hipFuncSetAttribute((const void*)fused_gemm<true>,
                              hipFuncAttributeMaxDynamicSharedMemorySize, SMEM_SYNC);
    fused_gemm<true><<<grid, 256, SMEM_SYNC, stream>>>(X, P, Tb, Tf, Out);
#endif
  } else {
    (void)hipFuncSetAttribute((const void*)fused_gemm<false>,
                              hipFuncAttributeMaxDynamicSharedMemorySize, SMEM_SYNC);
    fused_gemm<false><<<grid, 256, SMEM_SYNC, stream>>>(X, P, nullptr, Tf, Out);
  }
}
